// OptimusTXLDecoder_36455682408471
// MI455X (gfx1250) — compile-verified, matrix operand order checked by reading, not run on hardware
//
#include <hip/hip_runtime.h>
#include <cstdint>

// ---------------------------------------------------------------------------
// Types for CDNA5 WMMA (wave32, v_wmma_f32_16x16x32_f16)
// ---------------------------------------------------------------------------
typedef _Float16 h16;
typedef __attribute__((ext_vector_type(16))) _Float16 v16h;
typedef __attribute__((ext_vector_type(8)))  _Float16 v8h;
typedef __attribute__((ext_vector_type(8)))  float    v8f;

static constexpr int QLEN = 1024;
static constexpr int MLEN = 1024;
static constexpr int KLEN = 2048;   // QLEN + MLEN
static constexpr int BB   = 4;      // batch
static constexpr int D    = 1024;   // model dim
static constexpr int NH   = 16;
static constexpr int DH   = 64;
static constexpr int DI   = 4096;
static constexpr int HLD  = 3072;   // heads leading dim (3*NH*DH)

#define DEV static __device__ __forceinline__

// ---- gfx1250 async global->LDS path (ASYNCcnt), guarded by __has_builtin ----
#if defined(__has_builtin)
#if __has_builtin(__builtin_amdgcn_global_load_async_to_lds_b128) && \
    __has_builtin(__builtin_amdgcn_s_wait_asynccnt)
#define USE_ASYNC 1
#endif
#endif
#ifndef USE_ASYNC
#define USE_ASYNC 0
#endif

#if USE_ASYNC
// Builtin signature (from hipcc diagnostic): param0 is
// 'int __attribute__((vector_size(16))) __device__ *'  == AS(1) v4i*,
// param1 is the LDS destination (AS(3) v4i*), then imm offset + imm cpol.
typedef int v4i __attribute__((vector_size(16)));
typedef __attribute__((address_space(1))) v4i* gptr_v4i;
typedef __attribute__((address_space(3))) v4i* lptr_v4i;
DEV gptr_v4i to_glob(const void* p) { return (gptr_v4i)(uintptr_t)p; }
DEV lptr_v4i to_lds(void* p) { return (lptr_v4i)(uint32_t)(uintptr_t)p; }
#endif

DEV v8f zero8() {
  v8f z;
#pragma unroll
  for (int i = 0; i < 8; ++i) z[i] = 0.0f;
  return z;
}

DEV v8f wmma16(v16h a, v16h b, v8f c) {
  // D = A(16x32) * B(32x16) + C, f32 accumulate
  return __builtin_amdgcn_wmma_f32_16x16x32_f16(false, a, false, b, (short)0, c, false, false);
}

// Load a 16x32 f16 fragment (A-layout; B-fragments load identically from B^T).
// p points at (tile_row0, tile_col0); rs = row stride in elements.
// Lane layout (ISA 7.12.2, 16-bit A 16x32): lanes 0-15 hold K 0..7,16..23;
// lanes 16-31 hold K 8..15,24..31 of row M=lane&15.
DEV v16h load_frag(const h16* p, size_t rs, int lane) {
  const int r = lane & 15, hi = (lane >> 4) & 1;
  const h16* q = p + (size_t)r * rs + hi * 8;
  union { v16h v; v8h h[2]; } u;
  u.h[0] = *(const v8h*)q;
  u.h[1] = *(const v8h*)(q + 16);
  return u.v;
}

// Same, but clamps the row index (for rel-emb tiles that spill past KLEN; the
// spilled values land only in causally-masked score positions).
DEV v16h load_frag_clamp(const h16* p, size_t rs, int row0, int maxrow, int lane) {
  const int r = lane & 15, hi = (lane >> 4) & 1;
  int rr = row0 + r; if (rr > maxrow) rr = maxrow; if (rr < 0) rr = 0;
  const h16* q = p + (size_t)rr * rs + hi * 8;
  union { v16h v; v8h h[2]; } u;
  u.h[0] = *(const v8h*)q;
  u.h[1] = *(const v8h*)(q + 16);
  return u.v;
}

// ---------------------------------------------------------------------------
// Conversion / layout kernels
// ---------------------------------------------------------------------------
__global__ void k_build_cat(const float* __restrict__ dec, const float* __restrict__ mems,
                            h16* __restrict__ cat) {
  const size_t i = (size_t)blockIdx.x * blockDim.x + threadIdx.x;
  const size_t total = (size_t)KLEN * BB * D;
  if (i >= total) return;
  const size_t half = (size_t)MLEN * BB * D;
  const float v = (i < half) ? mems[i] : dec[i - half];
  cat[i] = (h16)v;
}

// W: K x N (f32, row-major) -> Wt: N x K (f16, row-major)
__global__ void k_transpose(const float* __restrict__ W, h16* __restrict__ Wt, int K, int N) {
  const size_t idx = (size_t)blockIdx.x * blockDim.x + threadIdx.x;
  if (idx >= (size_t)K * N) return;
  const int n = (int)(idx / K);
  const int k = (int)(idx % K);
  Wt[idx] = (h16)W[(size_t)k * N + n];
}

__global__ void k_convert(const float* __restrict__ src, h16* __restrict__ dst, size_t n) {
  const size_t i = (size_t)blockIdx.x * blockDim.x + threadIdx.x;
  if (i < n) dst[i] = (h16)src[i];
}

// qb[i*BB+b][c] = heads[(MLEN+i)*BB+b][c] + r_w_bias[c]   (c = n*DH+d)
__global__ void k_qb(const h16* __restrict__ heads, const float* __restrict__ rwb,
                     h16* __restrict__ qb) {
  const size_t idx = (size_t)blockIdx.x * blockDim.x + threadIdx.x;
  if (idx >= (size_t)QLEN * BB * D) return;
  const int c = (int)(idx & (D - 1));
  const size_t row = idx >> 10;          // i*BB + b
  const int i = (int)(row >> 2);
  const int b = (int)(row & 3);
  const float v = (float)heads[((size_t)(MLEN + i) * BB + b) * HLD + c] + rwb[c];
  qb[idx] = (h16)v;
}

// vT[b][n][d][j] = heads[j*BB+b][2048 + n*64 + d]
__global__ void k_vt(const h16* __restrict__ heads, h16* __restrict__ vT) {
  const size_t idx = (size_t)blockIdx.x * blockDim.x + threadIdx.x;
  if (idx >= (size_t)BB * NH * DH * KLEN) return;
  const int j = (int)(idx & (KLEN - 1));
  size_t t = idx >> 11;
  const int d = (int)(t & 63); t >>= 6;
  const int n = (int)(t & 15); t >>= 4;
  const int b = (int)t;
  vT[idx] = heads[((size_t)j * BB + b) * HLD + 2 * D + n * DH + d];
}

// ---------------------------------------------------------------------------
// Generic WMMA GEMM:  C[M x N] = A[M x K] * Bt[N x K]^T   (f16 in, f32 acc)
// block = 128 threads (4 waves), tile = 128 rows x 64 cols, wave = 32x64.
// B tile (shared by all 4 waves) is staged in LDS, double-buffered; on
// gfx1250 the fill uses GLOBAL_LOAD_ASYNC_TO_LDS_B128 + s_wait_asynccnt.
// A fragments stay as direct global b128 loads (per-wave, no reuse in block).
// ---------------------------------------------------------------------------
__global__ __launch_bounds__(128) void gemm_f16_k(
    const h16* __restrict__ A, int lda,
    const h16* __restrict__ Bt, int ldb,
    int K, int ldc,
    float* __restrict__ Cf, h16* __restrict__ Ch,
    const float* __restrict__ bias, int relu) {
  const int lane = threadIdx.x & 31;
  const int w = threadIdx.x >> 5;
  const int t = threadIdx.x;
  const int n0 = blockIdx.x * 64;
  const int m0 = blockIdx.y * 128 + w * 32;

  // 2 x (64 rows of B^T x 32 k) f16 tiles = 2 x 4KB
  __shared__ __align__(16) h16 bstage[2][64 * 32];

  // cooperative fill: 256 16B chunks per stage, 2 per thread
  const int c0 = t,        r0 = c0 >> 2, o0 = (c0 & 3) * 8;
  const int c1 = t + 128,  r1 = c1 >> 2, o1 = (c1 & 3) * 8;

  auto issueB = [&](int p, int k0) {
#if USE_ASYNC
    __builtin_amdgcn_global_load_async_to_lds_b128(
        to_glob(Bt + (size_t)(n0 + r0) * ldb + k0 + o0),
        to_lds(&bstage[p][r0 * 32 + o0]), 0, 0);
    __builtin_amdgcn_global_load_async_to_lds_b128(
        to_glob(Bt + (size_t)(n0 + r1) * ldb + k0 + o1),
        to_lds(&bstage[p][r1 * 32 + o1]), 0, 0);
#else
    *(v8h*)&bstage[p][r0 * 32 + o0] = *(const v8h*)(Bt + (size_t)(n0 + r0) * ldb + k0 + o0);
    *(v8h*)&bstage[p][r1 * 32 + o1] = *(const v8h*)(Bt + (size_t)(n0 + r1) * ldb + k0 + o1);
#endif
  };

  v8f acc[2][4];
#pragma unroll
  for (int i = 0; i < 2; ++i)
#pragma unroll
    for (int j = 0; j < 4; ++j) acc[i][j] = zero8();

  issueB(0, 0);
  int p = 0;
  for (int k0 = 0; k0 < K; k0 += 32, p ^= 1) {
    const bool pre = (k0 + 32) < K;
    if (pre) issueB(p ^ 1, k0 + 32);
#if USE_ASYNC
    if (pre) { __builtin_amdgcn_s_wait_asynccnt(2); }
    else     { __builtin_amdgcn_s_wait_asynccnt(0); }
#endif
    __syncthreads();   // stage p complete & visible to all waves

    v16h a0 = load_frag(A + (size_t)m0 * lda + k0, (size_t)lda, lane);
    v16h a1 = load_frag(A + (size_t)(m0 + 16) * lda + k0, (size_t)lda, lane);
#pragma unroll
    for (int nt = 0; nt < 4; ++nt) {
      v16h bf = load_frag(&bstage[p][nt * 16 * 32], 32, lane);
      acc[0][nt] = wmma16(a0, bf, acc[0][nt]);
      acc[1][nt] = wmma16(a1, bf, acc[1][nt]);
    }
    __syncthreads();   // done reading stage p before it is refilled
  }

  const int hi = lane >> 4, nn = lane & 15;
#pragma unroll
  for (int rt = 0; rt < 2; ++rt)
#pragma unroll
    for (int nt = 0; nt < 4; ++nt)
#pragma unroll
      for (int v = 0; v < 8; ++v) {
        const int row = m0 + rt * 16 + v + 8 * hi;
        const int col = n0 + nt * 16 + nn;
        float x = acc[rt][nt][v];
        if (bias) x += bias[col];
        if (relu) x = fmaxf(x, 0.0f);
        if (Cf) Cf[(size_t)row * ldc + col] = x;
        if (Ch) Ch[(size_t)row * ldc + col] = (h16)x;
      }
}

// ---------------------------------------------------------------------------
// Flash-style rel-attention. grid = (QLEN/64, NH, BB), block = 128 (4 waves).
// Wave w owns q rows [i0+16w, i0+16w+16). Streams 64-wide K tiles with online
// softmax. BD term recomputed inline: B_[i, j+QLEN-1-i] + r_bias[...], via a
// 16x96 rel-GEMM tile per wave per k-tile staged in LDS (covers the diagonal
// shift window), then gathered with col = (j-k0) + 15 - (i-iw0).
// ---------------------------------------------------------------------------
__global__ __launch_bounds__(128) void attn_f16_k(
    const h16* __restrict__ heads, const h16* __restrict__ qb,
    const h16* __restrict__ vT, const h16* __restrict__ remb,
    const float* __restrict__ r_bias, h16* __restrict__ att) {
  const int lane = threadIdx.x & 31;
  const int w = threadIdx.x >> 5;
  const int i0 = blockIdx.x * 64;
  const int n = blockIdx.y;
  const int b = blockIdx.z;
  const int iw0 = i0 + w * 16;
  const int hi = lane >> 4, nn = lane & 15;

  __shared__ __align__(16) float srel[4][16][96];   // per-wave rel-score windows
  __shared__ __align__(16) h16   sP[4][16][64];     // per-wave P tiles

  const size_t qstride = (size_t)BB * D;     // 4096
  const size_t hstride = (size_t)BB * HLD;   // 12288

  const h16* qbp = qb + ((size_t)iw0 * BB + b) * D + (size_t)n * DH;
  const h16* qrp = heads + ((size_t)(MLEN + iw0) * BB + b) * HLD + (size_t)n * DH;
  const v16h qb0 = load_frag(qbp, qstride, lane);
  const v16h qb1 = load_frag(qbp + 32, qstride, lane);
  const v16h qr0 = load_frag(qrp, hstride, lane);
  const v16h qr1 = load_frag(qrp + 32, hstride, lane);

  v8f oacc[4];
#pragma unroll
  for (int tt = 0; tt < 4; ++tt) oacc[tt] = zero8();
  float mrun[8], lrun[8];
#pragma unroll
  for (int v = 0; v < 8; ++v) { mrun[v] = -1e30f; lrun[v] = 0.0f; }

  int k_end = i0 + MLEN + 64;
  if (k_end > KLEN) k_end = KLEN;

  for (int k0 = 0; k0 < k_end; k0 += 64) {
    // ---- rel scores: B_[i, jr] for jr in [relbase, relbase+96) + r_bias ----
    const int relbase = k0 + QLEN - 16 - iw0;
#pragma unroll
    for (int nt = 0; nt < 6; ++nt) {
      const int jr0 = relbase + nt * 16;
      v16h r0 = load_frag_clamp(remb + (size_t)n * DH, (size_t)NH * DH, jr0, KLEN - 1, lane);
      v16h r1 = load_frag_clamp(remb + (size_t)n * DH + 32, (size_t)NH * DH, jr0, KLEN - 1, lane);
      v8f c = zero8();
      c = wmma16(qr0, r0, c);
      c = wmma16(qr1, r1, c);
#pragma unroll
      for (int v = 0; v < 8; ++v) {
        const int r = v + 8 * hi;
        const int col = nt * 16 + nn;
        int jr = relbase + col; if (jr > KLEN - 1) jr = KLEN - 1;
        srel[w][r][col] = c[v] + r_bias[(size_t)jr * NH + n];
      }
    }
    // ---- content scores: (q + r_w_bias) @ K^T ----
    v8f sac[4];
#pragma unroll
    for (int nt = 0; nt < 4; ++nt) {
      const int j0 = k0 + nt * 16;
      const h16* kp = heads + ((size_t)j0 * BB + b) * HLD + D + (size_t)n * DH;
      v16h kf0 = load_frag(kp, hstride, lane);
      v16h kf1 = load_frag(kp + 32, hstride, lane);
      v8f c = zero8();
      c = wmma16(qb0, kf0, c);
      c = wmma16(qb1, kf1, c);
      sac[nt] = c;
    }
    // ---- combine, mask, online softmax (row stats in regs, 16-lane reduce) --
    float pv[4][8];
    float corr[8];
#pragma unroll
    for (int v = 0; v < 8; ++v) {
      const int r = v + 8 * hi;
      const int i = iw0 + r;
      float sv[4];
      float mx = -1e30f;
#pragma unroll
      for (int nt = 0; nt < 4; ++nt) {
        const int jc = nt * 16 + nn;
        float s = (sac[nt][v] + srel[w][r][jc + 15 - r]) * 0.125f;
        if (k0 + jc > i + MLEN) s = -1e30f;   // causal mask
        sv[nt] = s;
        mx = fmaxf(mx, s);
      }
#pragma unroll
      for (int off = 8; off > 0; off >>= 1) mx = fmaxf(mx, __shfl_xor(mx, off, 32));
      const float newm = fmaxf(mrun[v], mx);
      const float cr = __expf(mrun[v] - newm);
      mrun[v] = newm;
      float rs = 0.0f;
#pragma unroll
      for (int nt = 0; nt < 4; ++nt) {
        const float pe = __expf(sv[nt] - newm);
        pv[nt][v] = pe;
        rs += pe;
      }
#pragma unroll
      for (int off = 8; off > 0; off >>= 1) rs += __shfl_xor(rs, off, 32);
      lrun[v] = lrun[v] * cr + rs;
      corr[v] = cr;
    }
#pragma unroll
    for (int dt = 0; dt < 4; ++dt)
#pragma unroll
      for (int v = 0; v < 8; ++v) oacc[dt][v] *= corr[v];
    // ---- P -> LDS (f16), then O += P @ V (B frags from pre-transposed V) ---
#pragma unroll
    for (int nt = 0; nt < 4; ++nt)
#pragma unroll
      for (int v = 0; v < 8; ++v)
        sP[w][v + 8 * hi][nt * 16 + nn] = (h16)pv[nt][v];
#pragma unroll
    for (int kk = 0; kk < 64; kk += 32) {
      v16h pa = load_frag(&sP[w][0][kk], 64, lane);
#pragma unroll
      for (int dt = 0; dt < 4; ++dt) {
        const h16* vp = vT + ((size_t)(b * NH + n) * DH + dt * 16) * KLEN + k0 + kk;
        v16h vb = load_frag(vp, (size_t)KLEN, lane);
        oacc[dt] = wmma16(pa, vb, oacc[dt]);
      }
    }
  }
  // ---- normalize and write attn_vec (f16) ----
#pragma unroll
  for (int dt = 0; dt < 4; ++dt)
#pragma unroll
    for (int v = 0; v < 8; ++v) {
      const int i = iw0 + v + 8 * hi;
      const float o = oacc[dt][v] / (lrun[v] * (1.0f + 1e-8f));
      att[((size_t)i * BB + b) * D + (size_t)n * DH + dt * 16 + nn] = (h16)o;
    }
}

// ---------------------------------------------------------------------------
// Row LayerNorm: out = LN(X + Y) * g + b ; optional f16 copy. One block/row.
// ---------------------------------------------------------------------------
__global__ __launch_bounds__(256) void ln_k(
    const float* __restrict__ X, const float* __restrict__ Y,
    const float* __restrict__ g, const float* __restrict__ be,
    float* __restrict__ outF, h16* __restrict__ outH) {
  const int row = blockIdx.x;
  const int t = threadIdx.x;
  const float* x = X + (size_t)row * D;
  const float* y = Y + (size_t)row * D;
  float vals[4];
  float s = 0.0f, s2 = 0.0f;
#pragma unroll
  for (int e = 0; e < 4; ++e) {
    const float v = x[t + e * 256] + y[t + e * 256];
    vals[e] = v; s += v; s2 += v * v;
  }
  __shared__ float sbuf[256], s2buf[256];
  sbuf[t] = s; s2buf[t] = s2;
  __syncthreads();
  for (int o = 128; o > 0; o >>= 1) {
    if (t < o) { sbuf[t] += sbuf[t + o]; s2buf[t] += s2buf[t + o]; }
    __syncthreads();
  }
  const float mean = sbuf[0] * (1.0f / D);
  const float var = s2buf[0] * (1.0f / D) - mean * mean;
  const float inv = rsqrtf(var + 1e-5f);
#pragma unroll
  for (int e = 0; e < 4; ++e) {
    const int c = t + e * 256;
    const float o = (vals[e] - mean) * inv * g[c] + be[c];
    if (outF) outF[(size_t)row * D + c] = o;
    if (outH) outH[(size_t)row * D + c] = (h16)o;
  }
}

// ---------------------------------------------------------------------------
// Host orchestration
// ---------------------------------------------------------------------------
extern "C" void kernel_launch(void* const* d_in, const int* in_sizes, int n_in,
                              void* d_out, int out_size, void* d_ws, size_t ws_size,
                              hipStream_t stream) {
  (void)in_sizes; (void)n_in; (void)out_size; (void)ws_size;
  const float* dec_inp  = (const float*)d_in[0];
  const float* mems     = (const float*)d_in[1];
  const float* r_emb    = (const float*)d_in[2];
  const float* r_w_bias = (const float*)d_in[3];
  const float* r_bias   = (const float*)d_in[4];
  const float* qkv_w    = (const float*)d_in[5];
  const float* o_w      = (const float*)d_in[6];
  const float* ln1_g    = (const float*)d_in[7];
  const float* ln1_b    = (const float*)d_in[8];
  const float* ff_w1    = (const float*)d_in[9];
  const float* ff_b1    = (const float*)d_in[10];
  const float* ff_w2    = (const float*)d_in[11];
  const float* ff_b2    = (const float*)d_in[12];
  const float* ln2_g    = (const float*)d_in[13];
  const float* ln2_b    = (const float*)d_in[14];
  // d_in[15] = dec_attn_mask (causal; applied analytically in-kernel)

  char* p = (char*)d_ws;
  auto take = [&](size_t bytes) -> char* {
    char* r = p; p += (bytes + 255) & ~(size_t)255; return r;
  };
  h16*   catH    = (h16*)take((size_t)KLEN * BB * D * 2);
  h16*   qkvT    = (h16*)take((size_t)HLD * D * 2);
  h16*   oT      = (h16*)take((size_t)D * D * 2);
  h16*   w1T     = (h16*)take((size_t)DI * D * 2);
  h16*   w2T     = (h16*)take((size_t)D * DI * 2);
  h16*   rembH   = (h16*)take((size_t)KLEN * NH * DH * 2);
  h16*   headsH  = (h16*)take((size_t)KLEN * BB * HLD * 2);
  h16*   qbH     = (h16*)take((size_t)QLEN * BB * D * 2);
  h16*   vTH     = (h16*)take((size_t)BB * NH * DH * KLEN * 2);
  h16*   attH    = (h16*)take((size_t)QLEN * BB * D * 2);
  float* attnOut = (float*)take((size_t)QLEN * BB * D * 4);
  float* out1F   = (float*)take((size_t)QLEN * BB * D * 4);
  h16*   out1H   = (h16*)take((size_t)QLEN * BB * D * 2);
  h16*   h1H     = (h16*)take((size_t)QLEN * BB * DI * 2);
  float* coreF   = (float*)take((size_t)QLEN * BB * D * 4);

  const int T = 256;
  auto nb = [&](size_t n) { return (unsigned)((n + T - 1) / T); };

  // Stage 0: precision/layout conversion
  k_build_cat<<<nb((size_t)KLEN * BB * D), T, 0, stream>>>(dec_inp, mems, catH);
  k_transpose<<<nb((size_t)D * HLD), T, 0, stream>>>(qkv_w, qkvT, D, HLD);
  k_transpose<<<nb((size_t)D * D), T, 0, stream>>>(o_w, oT, D, D);
  k_transpose<<<nb((size_t)D * DI), T, 0, stream>>>(ff_w1, w1T, D, DI);
  k_transpose<<<nb((size_t)DI * D), T, 0, stream>>>(ff_w2, w2T, DI, D);
  k_convert<<<nb((size_t)KLEN * NH * DH), T, 0, stream>>>(r_emb, rembH, (size_t)KLEN * NH * DH);

  // Stage 1: heads = cat @ qkv_w   (8192 x 1024 x 3072)
  gemm_f16_k<<<dim3(HLD / 64, (KLEN * BB) / 128), 128, 0, stream>>>(
      catH, D, qkvT, D, D, HLD, nullptr, headsH, nullptr, 0);

  // Stage 1b: biased q and transposed V
  k_qb<<<nb((size_t)QLEN * BB * D), T, 0, stream>>>(headsH, r_w_bias, qbH);
  k_vt<<<nb((size_t)BB * NH * DH * KLEN), T, 0, stream>>>(headsH, vTH);

  // Stage 2: flash rel-attention -> attn_vec (f16)
  attn_f16_k<<<dim3(QLEN / 64, NH, BB), 128, 0, stream>>>(
      headsH, qbH, vTH, rembH, r_bias, attH);

  // Stage 3: attn_out = attn_vec @ o_w (f32), then LN1(w + attn_out)
  gemm_f16_k<<<dim3(D / 64, (QLEN * BB) / 128), 128, 0, stream>>>(
      attH, D, oT, D, D, D, attnOut, nullptr, nullptr, 0);
  ln_k<<<QLEN * BB, 256, 0, stream>>>(dec_inp, attnOut, ln1_g, ln1_b, out1F, out1H);

  // Stage 4: FFN
  gemm_f16_k<<<dim3(DI / 64, (QLEN * BB) / 128), 128, 0, stream>>>(
      out1H, D, w1T, D, D, DI, nullptr, h1H, ff_b1, 1);          // relu(x@W1+b1)
  gemm_f16_k<<<dim3(D / 64, (QLEN * BB) / 128), 128, 0, stream>>>(
      h1H, DI, w2T, DI, DI, D, coreF, nullptr, ff_b2, 0);        // @W2+b2

  // Stage 5: LN2(out1 + core) -> d_out (f32)
  ln_k<<<QLEN * BB, 256, 0, stream>>>(out1F, coreF, ln2_g, ln2_b, (float*)d_out, nullptr);
}
